// EfficientNet_V2_face_63221918597379
// MI455X (gfx1250) — compile-verified, matrix operand order checked by reading, not run on hardware
//
#include <hip/hip_runtime.h>
#include <hip/hip_bf16.h>
#include <stdint.h>

// ---------------------------------------------------------------------------
// ArcFace margin head on MI455X (gfx1250, wave32, WMMA).
// out = s * blend(margin(cos), cos),  cos = normalize(x) @ normalize(W)^T
// fp32 inputs -> bf16 hi/lo split products on v_wmma_f32_16x16x32_bf16.
// Double-buffered LDS; A-tile staged via GLOBAL_LOAD_ASYNC_TO_LDS_B128
// (ASYNCcnt); L2-friendly dispatch (weight-tile sharers adjacent).
// ---------------------------------------------------------------------------

typedef __attribute__((ext_vector_type(16))) __bf16 v16bf;
typedef __attribute__((ext_vector_type(8)))  float  v8f;

#define BM 128
#define BN 128
#define BK 32
#define LDSS 40   // bf16 elements per LDS row (32 + 8 pad), 80B = 16B aligned

__device__ __constant__ float kCOS_M = 0.87758256189037271612f;  // cos(0.5)
__device__ __constant__ float kSIN_M = 0.47942553860420300538f;  // sin(0.5)
__device__ __constant__ float kTH    = -0.87758256189037271612f; // cos(pi-0.5)
__device__ __constant__ float kMM    = 0.23971276930210150269f;  // sin(pi-0.5)*0.5
__device__ __constant__ float kS     = 64.0f;

// Split an f32 into bf16 high part (RNE) + bf16 residual.
__device__ __forceinline__ void bf16split(float v, unsigned short& h, unsigned short& l) {
    unsigned int u = __float_as_uint(v);
    unsigned int r = u + 0x7FFFu + ((u >> 16) & 1u);
    h = (unsigned short)(r >> 16);
    float hf = __uint_as_float(((unsigned int)h) << 16);
    unsigned int u2 = __float_as_uint(v - hf);
    unsigned int r2 = u2 + 0x7FFFu + ((u2 >> 16) & 1u);
    l = (unsigned short)(r2 >> 16);
}

// Async copy 16B: global -> LDS, tracked by ASYNCcnt (gfx1250).
// IOFFSET is added to both the LDS and global addresses (ISA 08 §4.4).
__device__ __forceinline__ void async_copy16(unsigned int lds_addr, unsigned long long gaddr) {
    asm volatile("global_load_async_to_lds_b128 %0, %1, off"
                 :: "v"(lds_addr), "v"(gaddr) : "memory");
}
__device__ __forceinline__ void async_copy16_off16(unsigned int lds_addr, unsigned long long gaddr) {
    asm volatile("global_load_async_to_lds_b128 %0, %1, off offset:16"
                 :: "v"(lds_addr), "v"(gaddr) : "memory");
}
__device__ __forceinline__ void wait_async0() {
    asm volatile("s_wait_asynccnt 0" ::: "memory");
}
// Low 32 bits of a generic shared-aperture pointer == LDS byte address.
__device__ __forceinline__ unsigned int lds_addr_of(const void* p) {
    return (unsigned int)(unsigned long long)(uintptr_t)p;
}

// --------------------------- x: normalize + split ---------------------------
__global__ void xnorm_split_kernel(const float* __restrict__ x,
                                   unsigned short* __restrict__ xh,
                                   unsigned short* __restrict__ xl,
                                   int D) {
    __shared__ float red[256];
    const int row = blockIdx.x;
    const int t   = threadIdx.x;
    const float* xr = x + (size_t)row * D;
    float s = 0.0f;
    for (int i = t; i < D; i += 256) { float v = xr[i]; s += v * v; }
    red[t] = s;
    __syncthreads();
    for (int off = 128; off > 0; off >>= 1) {
        if (t < off) red[t] += red[t + off];
        __syncthreads();
    }
    const float inv = 1.0f / fmaxf(__builtin_sqrtf(red[0]), 1e-12f);
    for (int i = t; i < D; i += 256) {
        unsigned short h, l;
        bf16split(xr[i] * inv, h, l);
        xh[(size_t)row * D + i] = h;
        xl[(size_t)row * D + i] = l;
    }
}

// --------------------------- W: inverse row norms ---------------------------
__global__ void winv_kernel(const float* __restrict__ w,
                            float* __restrict__ winv, int C, int D) {
    const int lane = threadIdx.x & 31;
    const int wid  = threadIdx.x >> 5;
    const int row  = blockIdx.x * 8 + wid;
    if (row >= C) return;
    const float4* wr = (const float4*)(w + (size_t)row * D);
    float s = 0.0f;
    for (int i = lane; i < (D >> 2); i += 32) {
        float4 f = wr[i];
        s += f.x * f.x + f.y * f.y + f.z * f.z + f.w * f.w;
    }
    #pragma unroll
    for (int off = 16; off > 0; off >>= 1) s += __shfl_xor(s, off, 32);
    if (lane == 0) winv[row] = 1.0f / fmaxf(__builtin_sqrtf(s), 1e-12f);
}

// ------------------------------- main GEMM ----------------------------------
__global__ __launch_bounds__(256, 2)
void arcface_gemm_kernel(const unsigned short* __restrict__ xh,
                         const unsigned short* __restrict__ xl,
                         const float* __restrict__ weight,
                         const float* __restrict__ winv,
                         const int* __restrict__ label,
                         float* __restrict__ out,
                         int Bsz, int D, int C) {
    // Double-buffered tiles: 2 * 4 * 128*40*2B = 80 KB LDS.
    __shared__ __align__(16) unsigned short lds_ah[2][BM * LDSS];
    __shared__ __align__(16) unsigned short lds_al[2][BM * LDSS];
    __shared__ __align__(16) unsigned short lds_wh[2][BN * LDSS];
    __shared__ __align__(16) unsigned short lds_wl[2][BN * LDSS];

    const int tid   = threadIdx.x;
    const int lane  = tid & 31;
    const int wid   = tid >> 5;     // 8 waves
    const int wm    = wid >> 2;     // 0..1  (64 rows each)
    const int wn    = wid & 3;      // 0..3  (32 cols each)
    const int mlane = lane & 15;
    const int khalf = lane >> 4;    // wave32: lanes 16-31 hold the other K half

    // blockIdx.x (fast) = M blocks -> weight-tile sharers dispatch adjacently.
    const int m0 = blockIdx.x * BM;
    const int n0 = blockIdx.y * BN;

    // loader coords: 256 threads -> 128 rows x 2 halves of 16 elements
    const int lr = tid >> 1;
    const int lh = tid & 1;
    int am = m0 + lr; if (am >= Bsz) am = Bsz - 1;
    int wr = n0 + lr; if (wr >= C)   wr = C - 1;
    const float winvr = winv[wr];

    const unsigned int ldsoff = (unsigned int)(lr * LDSS + lh * 16);

    v8f acc[4][2];
    #pragma unroll
    for (int mt = 0; mt < 4; ++mt)
        #pragma unroll
        for (int nt = 0; nt < 2; ++nt)
            #pragma unroll
            for (int r = 0; r < 8; ++r) acc[mt][nt][r] = 0.0f;

    union FragU { v16bf v; uint4 q[2]; };

    // A tile: async global->LDS, no VGPR staging.
    auto a_tile_async = [&](int buf, int k0) {
        const size_t aoff = (size_t)am * D + k0 + lh * 16;
        unsigned int dh = lds_addr_of(&lds_ah[buf][ldsoff]);
        unsigned int dl = lds_addr_of(&lds_al[buf][ldsoff]);
        unsigned long long gh = (unsigned long long)(uintptr_t)(xh + aoff);
        unsigned long long gl = (unsigned long long)(uintptr_t)(xl + aoff);
        async_copy16(dh, gh);  async_copy16_off16(dh, gh);
        async_copy16(dl, gl);  async_copy16_off16(dl, gl);
    };

    // W tile: raw fp32 held in regs across the WMMA block.
    float4 wreg[4];
    auto w_tile_load = [&](int k0) {
        const float* srcW = weight + (size_t)wr * D + k0 + lh * 16;
        __builtin_prefetch(srcW + 2 * BK, 0, 0);   // 2 tiles ahead
        #pragma unroll
        for (int i = 0; i < 4; ++i) wreg[i] = *(const float4*)(srcW + i * 4);
    };
    auto w_tile_store = [&](int buf) {
        unsigned int hbuf[8], lbuf[8];
        #pragma unroll
        for (int i = 0; i < 4; ++i) {
            unsigned short h0, l0, h1, l1, h2, l2, h3, l3;
            bf16split(wreg[i].x * winvr, h0, l0);
            bf16split(wreg[i].y * winvr, h1, l1);
            bf16split(wreg[i].z * winvr, h2, l2);
            bf16split(wreg[i].w * winvr, h3, l3);
            hbuf[i * 2 + 0] = (unsigned)h0 | ((unsigned)h1 << 16);
            hbuf[i * 2 + 1] = (unsigned)h2 | ((unsigned)h3 << 16);
            lbuf[i * 2 + 0] = (unsigned)l0 | ((unsigned)l1 << 16);
            lbuf[i * 2 + 1] = (unsigned)l2 | ((unsigned)l3 << 16);
        }
        uint4* dwh = (uint4*)&lds_wh[buf][ldsoff];
        uint4* dwl = (uint4*)&lds_wl[buf][ldsoff];
        dwh[0] = *(uint4*)&hbuf[0]; dwh[1] = *(uint4*)&hbuf[4];
        dwl[0] = *(uint4*)&lbuf[0]; dwl[1] = *(uint4*)&lbuf[4];
    };

    // Prologue: stage tile 0 into buffer 0.
    a_tile_async(0, 0);
    w_tile_load(0);
    w_tile_store(0);

    int cur = 0;
    for (int k0 = 0; k0 < D; k0 += BK) {
        wait_async0();                   // drain this wave's async LDS writes
        __syncthreads();                 // buf `cur` ready; buf `cur^1` free
        const bool have_next = (k0 + BK < D);   // uniform branch, EXEC stays full
        if (have_next) {
            a_tile_async(cur ^ 1, k0 + BK);     // async: no regs held
            w_tile_load(k0 + BK);               // 16 VGPRs held across WMMA
        }

        // ---- B fragments hoisted (4 x 8 VGPRs) ----
        FragU bhf[2], blf[2];
        #pragma unroll
        for (int nt = 0; nt < 2; ++nt) {
            const int row = wn * 32 + nt * 16 + mlane;
            const unsigned short* ph = &lds_wh[cur][row * LDSS];
            const unsigned short* pl = &lds_wl[cur][row * LDSS];
            bhf[nt].q[0] = *(const uint4*)(ph + khalf * 8);
            bhf[nt].q[1] = *(const uint4*)(ph + 16 + khalf * 8);
            blf[nt].q[0] = *(const uint4*)(pl + khalf * 8);
            blf[nt].q[1] = *(const uint4*)(pl + 16 + khalf * 8);
        }

        // ---- A fragments per mt (register reuse keeps pressure low) ----
        #pragma unroll
        for (int mt = 0; mt < 4; ++mt) {
            const int row = wm * 64 + mt * 16 + mlane;
            const unsigned short* ph = &lds_ah[cur][row * LDSS];
            const unsigned short* pl = &lds_al[cur][row * LDSS];
            FragU ahf, alf;
            ahf.q[0] = *(const uint4*)(ph + khalf * 8);
            ahf.q[1] = *(const uint4*)(ph + 16 + khalf * 8);
            alf.q[0] = *(const uint4*)(pl + khalf * 8);
            alf.q[1] = *(const uint4*)(pl + 16 + khalf * 8);
            #pragma unroll
            for (int nt = 0; nt < 2; ++nt) {
                acc[mt][nt] = __builtin_amdgcn_wmma_f32_16x16x32_bf16(
                    false, ahf.v, false, bhf[nt].v, (short)0, acc[mt][nt], false, false);
                acc[mt][nt] = __builtin_amdgcn_wmma_f32_16x16x32_bf16(
                    false, ahf.v, false, blf[nt].v, (short)0, acc[mt][nt], false, false);
                acc[mt][nt] = __builtin_amdgcn_wmma_f32_16x16x32_bf16(
                    false, alf.v, false, bhf[nt].v, (short)0, acc[mt][nt], false, false);
            }
        }

        if (have_next) w_tile_store(cur ^ 1);   // convert + ds_store behind WMMA
        cur ^= 1;
    }

    // ---- ArcFace epilogue: C/D layout -> VGPR r: M = r + 8*khalf, N = mlane ----
    const int nbase = n0 + wn * 32;
    #pragma unroll
    for (int mt = 0; mt < 4; ++mt) {
        const int mbase = m0 + wm * 64 + mt * 16 + khalf * 8;
        #pragma unroll
        for (int nt = 0; nt < 2; ++nt) {
            const int n = nbase + nt * 16 + mlane;
            if (n < C) {
                #pragma unroll
                for (int r = 0; r < 8; ++r) {
                    const int m = mbase + r;
                    if (m < Bsz) {
                        float cosv = acc[mt][nt][r];
                        float s2   = fmaxf(1.0f - cosv * cosv, 0.0f);
                        float sinv = __builtin_sqrtf(s2);
                        float phi  = cosv * kCOS_M - sinv * kSIN_M;
                        phi        = (cosv > kTH) ? phi : (cosv - kMM);
                        bool  tgt  = (label[m] == n);
                        out[(size_t)m * C + n] = (tgt ? phi : cosv) * kS;
                    }
                }
            }
        }
    }
}

// ------------------------------- launcher -----------------------------------
extern "C" void kernel_launch(void* const* d_in, const int* in_sizes, int n_in,
                              void* d_out, int out_size, void* d_ws, size_t ws_size,
                              hipStream_t stream) {
    const float* x      = (const float*)d_in[0];
    const float* weight = (const float*)d_in[1];
    const int*   label  = (const int*)d_in[2];
    float*       out    = (float*)d_out;

    const int Bsz = in_sizes[2];            // 512
    const int D   = in_sizes[0] / Bsz;      // 512
    const int C   = in_sizes[1] / D;        // 100000

    char* ws = (char*)d_ws;
    size_t xBytes = ((size_t)Bsz * D * sizeof(unsigned short) + 255) & ~(size_t)255;
    unsigned short* xh   = (unsigned short*)ws;
    unsigned short* xl   = (unsigned short*)(ws + xBytes);
    float*          winv = (float*)(ws + 2 * xBytes);

    xnorm_split_kernel<<<Bsz, 256, 0, stream>>>(x, xh, xl, D);
    winv_kernel<<<(C + 7) / 8, 256, 0, stream>>>(weight, winv, C, D);

    // x fast-moving over M so weight-tile sharers are adjacent in dispatch.
    dim3 grid((Bsz + BM - 1) / BM, (C + BN - 1) / BN);
    arcface_gemm_kernel<<<grid, 256, 0, stream>>>(xh, xl, weight, winv, label, out,
                                                  Bsz, D, C);
}
